// Block_56075093016878
// MI455X (gfx1250) — matrix operand order repaired and checked by reading, not compiled, not run on hardware
//
#include <hip/hip_runtime.h>
#include <hip/hip_bf16.h>

#define DIM   384
#define HEADS 12
#define HD    32
#define NTOK  64
#define MLPH  1536
#define EPS   1e-5f

typedef __attribute__((ext_vector_type(16))) _Float16 v16h;
typedef __attribute__((ext_vector_type(8)))  _Float16 v8h;
typedef __attribute__((ext_vector_type(8)))  float    v8f;

__device__ __forceinline__ v8f wmma16(v16h a, v16h b, v8f c) {
  // D = A(16x32 f16) * B(32x16 f16) + C(16x16 f32)
  return __builtin_amdgcn_wmma_f32_16x16x32_f16(false, a, false, b, (short)0, c,
                                                false, false);
}

// A-fragment: lane<16 -> row m, K = {k0..k0+15}
//             lane>=16 -> row m, K = {k0+16..k0+31}
// caller passes p = &A[row*lda + k0 + hi*16]
__device__ __forceinline__ v16h make_a(const _Float16* p) {
  v8h x0 = *(const v8h*)p;
  v8h x1 = *(const v8h*)(p + 8);
  v16h r;
#pragma unroll
  for (int i = 0; i < 8; ++i) { r[i] = x0[i]; r[8 + i] = x1[i]; }
  return r;
}

// gfx1250 async copy: 16B global -> LDS, tracked by ASYNCcnt
__device__ __forceinline__ void async_b128(unsigned lds_addr, const void* gptr) {
  unsigned long long ga = (unsigned long long)(size_t)gptr;
  asm volatile("global_load_async_to_lds_b128 %0, %1, off"
               :: "v"(lds_addr), "v"(ga) : "memory");
}
__device__ __forceinline__ void wait_async0() {
  asm volatile("s_wait_asynccnt 0x0" ::: "memory");
}

// ---------------- weight cast f32 -> f16 ----------------
__global__ void cast_f32_f16(const float* __restrict__ s, _Float16* __restrict__ d, int n) {
  for (int i = blockIdx.x * blockDim.x + threadIdx.x; i < n; i += gridDim.x * blockDim.x)
    d[i] = (_Float16)s[i];
}

// ---------------- LayerNorm: one wave per row of 384 ----------------
__global__ __launch_bounds__(256) void ln_rows(const float* __restrict__ x,
                                               const float* __restrict__ g,
                                               const float* __restrict__ b,
                                               _Float16* __restrict__ out, int M) {
  int lane = threadIdx.x & 31;
  int wave = threadIdx.x >> 5;
  int row = blockIdx.x * 8 + wave;
  if (row >= M) return;
  const float* xr = x + (size_t)row * DIM;
  float v[12];
  float s = 0.f;
#pragma unroll
  for (int i = 0; i < 12; ++i) { v[i] = xr[lane + 32 * i]; s += v[i]; }
#pragma unroll
  for (int off = 16; off > 0; off >>= 1) s += __shfl_xor(s, off, 32);
  float mu = s * (1.0f / DIM);
  float q = 0.f;
#pragma unroll
  for (int i = 0; i < 12; ++i) { float d0 = v[i] - mu; q += d0 * d0; }
#pragma unroll
  for (int off = 16; off > 0; off >>= 1) q += __shfl_xor(q, off, 32);
  float rs = rsqrtf(q * (1.0f / DIM) + EPS);
  _Float16* orow = out + (size_t)row * DIM;
#pragma unroll
  for (int i = 0; i < 12; ++i) {
    int c = lane + 32 * i;
    orow[c] = (_Float16)((v[i] - mu) * rs * g[c] + b[c]);
  }
}

// ---------------- generic WMMA GEMM: out = A(MxK f16) @ W(NxK f16)^T + bias ----------------
// A tile (128 x 32) is async-DMA'd into LDS with double buffering; W read direct
// (all weights are L2-resident on a 192MB L2).
// MODE 1: scale cols < DIM by 1/sqrt(HD), write f16   (QKV)
// MODE 2: ReLU6, write f16                            (FC1)
// MODE 3: +resid, write f32                           (proj / FC2)
template <int MODE>
__global__ __launch_bounds__(256) void gemm_wmma(const _Float16* __restrict__ A,
                                                 const _Float16* __restrict__ W,
                                                 const float* __restrict__ bias,
                                                 const float* __restrict__ resid,
                                                 _Float16* __restrict__ outH,
                                                 float* __restrict__ outF,
                                                 int M, int N, int K) {
  __shared__ alignas(16) _Float16 Abuf[2][128 * 32];

  int tid = threadIdx.x;
  int lane = tid & 31;
  int wave = tid >> 5;
  int wm = wave & 3;   // 4 waves along M
  int wn = wave >> 2;  // 2 waves along N
  int blockM = blockIdx.y * 128;
  int tileM = blockM + wm * 32;
  int tileN = blockIdx.x * 64 + wn * 32;
  int l = lane & 15;
  int hi = lane >> 4;

  // per-thread async chunks: 2 x 16B per K-step (8KB tile / 256 threads)
  int ar0 = tid >> 2,          ao0 = (tid & 3) * 8;         // rows 0..63
  int ar1 = (tid + 256) >> 2,  ao1 = (tid & 3) * 8;         // rows 64..127
  const _Float16* ga0 = A + (size_t)(blockM + ar0) * K + ao0;
  const _Float16* ga1 = A + (size_t)(blockM + ar1) * K + ao1;
  unsigned la0[2], la1[2];
#pragma unroll
  for (int bb = 0; bb < 2; ++bb) {
    la0[bb] = (unsigned)(size_t)&Abuf[bb][ar0 * 32 + ao0];
    la1[bb] = (unsigned)(size_t)&Abuf[bb][ar1 * 32 + ao1];
  }

  const _Float16* W0 = W + (size_t)(tileN + l) * K + hi * 16;
  const _Float16* W1 = W + (size_t)(tileN + 16 + l) * K + hi * 16;

  v8f acc[2][2] = {};
  const int nk = K >> 5;

  // prologue: stage K-step 0 into buffer 0
  async_b128(la0[0], ga0);
  async_b128(la1[0], ga1);

  for (int ks = 0; ks < nk; ++ks) {
    int cur = ks & 1;
    wait_async0();        // own async loads for buffer `cur` complete
    __syncthreads();      // all waves' loads complete; safe to read cur / overwrite cur^1
    if (ks + 1 < nk) {
      int kn = (ks + 1) << 5;
      async_b128(la0[cur ^ 1], ga0 + kn);
      async_b128(la1[cur ^ 1], ga1 + kn);
    }
    int k0 = ks << 5;
    __builtin_prefetch(W0 + k0 + 256, 0, 0);
    v16h a0 = make_a(&Abuf[cur][(wm * 32 + l) * 32 + hi * 16]);
    v16h a1 = make_a(&Abuf[cur][(wm * 32 + 16 + l) * 32 + hi * 16]);
    v16h b0 = *(const v16h*)(W0 + k0);
    v16h b1 = *(const v16h*)(W1 + k0);
    acc[0][0] = wmma16(a0, b0, acc[0][0]);
    acc[0][1] = wmma16(a0, b1, acc[0][1]);
    acc[1][0] = wmma16(a1, b0, acc[1][0]);
    acc[1][1] = wmma16(a1, b1, acc[1][1]);
  }

#pragma unroll
  for (int mt = 0; mt < 2; ++mt) {
#pragma unroll
    for (int nt = 0; nt < 2; ++nt) {
      int col = tileN + nt * 16 + l;
      float bv = bias[col];
#pragma unroll
      for (int r = 0; r < 8; ++r) {
        size_t row = tileM + mt * 16 + hi * 8 + r;
        float v = acc[mt][nt][r] + bv;
        if (MODE == 1) { if (col < DIM) v *= 0.17677669529663687f; }
        if (MODE == 2) { v = fminf(fmaxf(v, 0.0f), 6.0f); }
        if (MODE == 3) outF[row * N + col] = v + resid[row * N + col];
        else           outH[row * N + col] = (_Float16)v;
      }
    }
  }
}

// ---------------- windowed attention: one wave per (window, head) ----------------
__global__ __launch_bounds__(32) void attn_win(const _Float16* __restrict__ qkv,
                                               const float* __restrict__ rpb,
                                               _Float16* __restrict__ o16) {
  __shared__ alignas(32) _Float16 Klds[NTOK * HD];   // [token][d]
  __shared__ alignas(32) _Float16 Vc[HD * NTOK];     // [d][token]  (transposed V)
  __shared__ alignas(32) _Float16 Plds[NTOK * NTOK]; // [m][n] softmax probs

  int lane = threadIdx.x;
  int l = lane & 15;
  int hi = lane >> 4;
  int win = blockIdx.x;
  int head = blockIdx.y;
  int bb = win >> 6;
  int wi = (win >> 3) & 7;
  int wj = win & 7;

  // token n (0..63) in this window -> global row
  auto grow = [&](int n) -> size_t {
    int hh = wi * 8 + (n >> 3);
    int ww = wj * 8 + (n & 7);
    return (size_t)bb * 4096 + (size_t)hh * 64 + ww;
  };

  // async-stage K tile (64 rows x 32 halves); 8 x b128 per wave
#pragma unroll
  for (int i = 0; i < 8; ++i) {
    int c = lane + 32 * i;
    int row = c >> 2;
    int off = (c & 3) * 8;
    size_t g = grow(row);
    async_b128((unsigned)(size_t)&Klds[row * HD + off],
               qkv + g * (3 * DIM) + DIM + head * HD + off);
  }
  // stage V transposed (manual: async DMA cannot transpose)
  for (int t = lane; t < NTOK; t += 32) {
    size_t g = grow(t);
    const _Float16* vp = qkv + g * (3 * DIM) + 2 * DIM + head * HD;
    v8h vv0 = *(const v8h*)vp;
    v8h vv1 = *(const v8h*)(vp + 8);
    v8h vv2 = *(const v8h*)(vp + 16);
    v8h vv3 = *(const v8h*)(vp + 24);
#pragma unroll
    for (int d = 0; d < 8; ++d) {
      Vc[d * NTOK + t] = vv0[d];
      Vc[(d + 8) * NTOK + t] = vv1[d];
      Vc[(d + 16) * NTOK + t] = vv2[d];
      Vc[(d + 24) * NTOK + t] = vv3[d];
    }
  }
  wait_async0();
  asm volatile("s_wait_dscnt 0x0" ::: "memory");
  __syncthreads();

  // S = Q * K^T  (64x64, 4x4 tiles of 16x16, K=32 -> one WMMA each)
  v8f S[4][4] = {};
#pragma unroll
  for (int i = 0; i < 4; ++i) {
    size_t gq = grow(i * 16 + l);
    v16h qf = make_a(qkv + gq * (3 * DIM) + head * HD + hi * 16);
#pragma unroll
    for (int j = 0; j < 4; ++j) {
      v16h kf = *(const v16h*)(Klds + (j * 16 + l) * HD + hi * 16);
      S[i][j] = wmma16(qf, kf, S[i][j]);
    }
  }

  // bias + softmax over rows (row m lives in one 16-lane half: shfl width 16)
#pragma unroll
  for (int i = 0; i < 4; ++i) {
#pragma unroll
    for (int r = 0; r < 8; ++r) {
      int m = i * 16 + hi * 8 + r;
#pragma unroll
      for (int j = 0; j < 4; ++j) {
        int n = j * 16 + l;
        int d0 = (m >> 3) - (n >> 3) + 7;
        int d1 = (m & 7) - (n & 7) + 7;
        S[i][j][r] += rpb[(d0 * 15 + d1) * HEADS + head];
      }
      float mx = fmaxf(fmaxf(S[i][0][r], S[i][1][r]), fmaxf(S[i][2][r], S[i][3][r]));
#pragma unroll
      for (int off = 8; off > 0; off >>= 1) mx = fmaxf(mx, __shfl_xor(mx, off, 16));
      float sum = 0.f;
#pragma unroll
      for (int j = 0; j < 4; ++j) {
        float e = __expf(S[i][j][r] - mx);
        S[i][j][r] = e;
        sum += e;
      }
#pragma unroll
      for (int off = 8; off > 0; off >>= 1) sum += __shfl_xor(sum, off, 16);
      float inv = 1.0f / sum;
#pragma unroll
      for (int j = 0; j < 4; ++j)
        Plds[m * NTOK + j * 16 + l] = (_Float16)(S[i][j][r] * inv);
    }
  }
  asm volatile("s_wait_dscnt 0x0" ::: "memory");
  __syncthreads();

  // O = P(64x64) * V(64x32): 4x2 tiles, K=64 -> two WMMA steps
  v8f O[4][2] = {};
#pragma unroll
  for (int i = 0; i < 4; ++i) {
#pragma unroll
    for (int ks = 0; ks < 2; ++ks) {
      int k0 = ks * 32;
      v16h pa = make_a(Plds + (i * 16 + l) * NTOK + k0 + hi * 16);
#pragma unroll
      for (int dt = 0; dt < 2; ++dt) {
        v16h vb = *(const v16h*)(Vc + (dt * 16 + l) * NTOK + k0 + hi * 16);
        O[i][dt] = wmma16(pa, vb, O[i][dt]);
      }
    }
  }

  // scatter back in original token order (window-reverse is implicit via grow())
#pragma unroll
  for (int i = 0; i < 4; ++i) {
#pragma unroll
    for (int r = 0; r < 8; ++r) {
      size_t g = grow(i * 16 + hi * 8 + r);
#pragma unroll
      for (int dt = 0; dt < 2; ++dt)
        o16[g * DIM + head * HD + dt * 16 + l] = (_Float16)O[i][dt][r];
    }
  }
}

// ---------------- host launch ----------------
extern "C" void kernel_launch(void* const* d_in, const int* in_sizes, int n_in,
                              void* d_out, int out_size, void* d_ws, size_t ws_size,
                              hipStream_t stream) {
  const float* x      = (const float*)d_in[0];
  const float* ln1_g  = (const float*)d_in[1];
  const float* ln1_b  = (const float*)d_in[2];
  const float* qkv_w  = (const float*)d_in[3];
  const float* qkv_b  = (const float*)d_in[4];
  const float* proj_w = (const float*)d_in[5];
  const float* proj_b = (const float*)d_in[6];
  const float* rpb    = (const float*)d_in[7];
  const float* ln2_g  = (const float*)d_in[8];
  const float* ln2_b  = (const float*)d_in[9];
  const float* fc1_w  = (const float*)d_in[10];
  const float* fc1_b  = (const float*)d_in[11];
  const float* fc2_w  = (const float*)d_in[12];
  const float* fc2_b  = (const float*)d_in[13];

  const int M = in_sizes[0] / DIM;  // B*L = 65536

  char* ws = (char*)d_ws;
  size_t off = 0;
  auto alloc = [&](size_t bytes) -> void* {
    off = (off + 255) & ~(size_t)255;
    void* p = ws + off;
    off += bytes;
    return p;
  };
  _Float16* wq16 = (_Float16*)alloc((size_t)3 * DIM * DIM * 2);
  _Float16* wp16 = (_Float16*)alloc((size_t)DIM * DIM * 2);
  _Float16* w1_16 = (_Float16*)alloc((size_t)MLPH * DIM * 2);
  _Float16* w2_16 = (_Float16*)alloc((size_t)DIM * MLPH * 2);
  _Float16* lnbuf = (_Float16*)alloc((size_t)M * DIM * 2);   // ln1, later ln2
  _Float16* big   = (_Float16*)alloc((size_t)M * MLPH * 2);  // qkv (M*1152), later mlp (M*1536)
  _Float16* o16   = (_Float16*)alloc((size_t)M * DIM * 2);
  float*    x1    = (float*)alloc((size_t)M * DIM * 4);

  // 1) cast weights to f16
  cast_f32_f16<<<dim3(512), dim3(256), 0, stream>>>(qkv_w, wq16, 3 * DIM * DIM);
  cast_f32_f16<<<dim3(256), dim3(256), 0, stream>>>(proj_w, wp16, DIM * DIM);
  cast_f32_f16<<<dim3(512), dim3(256), 0, stream>>>(fc1_w, w1_16, MLPH * DIM);
  cast_f32_f16<<<dim3(512), dim3(256), 0, stream>>>(fc2_w, w2_16, DIM * MLPH);

  // 2) LN1
  ln_rows<<<dim3(M / 8), dim3(256), 0, stream>>>(x, ln1_g, ln1_b, lnbuf, M);

  // 3) QKV GEMM (q pre-scaled by 1/sqrt(hd))
  gemm_wmma<1><<<dim3(3 * DIM / 64, M / 128), dim3(256), 0, stream>>>(
      lnbuf, wq16, qkv_b, nullptr, big, nullptr, M, 3 * DIM, DIM);

  // 4) windowed attention
  attn_win<<<dim3(M / NTOK, HEADS), dim3(32), 0, stream>>>(big, rpb, o16);

  // 5) proj + residual -> x1 (f32)
  gemm_wmma<3><<<dim3(DIM / 64, M / 128), dim3(256), 0, stream>>>(
      o16, wp16, proj_b, x, nullptr, x1, M, DIM, DIM);

  // 6) LN2
  ln_rows<<<dim3(M / 8), dim3(256), 0, stream>>>(x1, ln2_g, ln2_b, lnbuf, M);

  // 7) FC1 + ReLU6 -> big (f16, M x 1536)
  gemm_wmma<2><<<dim3(MLPH / 64, M / 128), dim3(256), 0, stream>>>(
      lnbuf, w1_16, fc1_b, nullptr, big, nullptr, M, MLPH, DIM);

  // 8) FC2 + residual -> d_out (f32)
  gemm_wmma<3><<<dim3(DIM / 64, M / 128), dim3(256), 0, stream>>>(
      big, w2_16, fc2_b, x1, nullptr, (float*)d_out, M, DIM, MLPH);
}
